// HFOpenMoeTop2Router_64089501991564
// MI455X (gfx1250) — matrix operand order repaired and checked by reading, not compiled, not run on hardware
//
#include <hip/hip_runtime.h>
#include <hip/hip_bf16.h>

typedef __attribute__((ext_vector_type(16))) _Float16 v16h;
typedef __attribute__((ext_vector_type(8)))  float    v8f;
typedef __attribute__((ext_vector_type(4)))  float    vf4;

#define T_TOK 8192
#define NEXP  64
#define CAP   320            // floor(2 * 1.25 * 8192 / 64) = 320, already even
#define NTILE (T_TOK / 32)   // 256 tiles of 32 tokens

// ---------------------------------------------------------------------------
// K1: per-token softmax + top-2. One wave32 per token; 2 logits per lane.
// ---------------------------------------------------------------------------
__global__ void topk_kernel(const float* __restrict__ x,
                            int* __restrict__ top1, int* __restrict__ top2,
                            float* __restrict__ w1, float* __restrict__ w2) {
    const int gtid = blockIdx.x * blockDim.x + threadIdx.x;
    const int tok  = gtid >> 5;
    const int lane = threadIdx.x & 31;

    const float2 v = reinterpret_cast<const float2*>(x + (size_t)tok * NEXP)[lane];
    float x0 = v.x, x1 = v.y;
    const int i0 = 2 * lane, i1 = 2 * lane + 1;

    // row max (softmax stabilization)
    float mx = fmaxf(x0, x1);
    #pragma unroll
    for (int off = 16; off; off >>= 1) mx = fmaxf(mx, __shfl_xor(mx, off, 32));

    // sum of exp
    float s = __expf(x0 - mx) + __expf(x1 - mx);
    #pragma unroll
    for (int off = 16; off; off >>= 1) s += __shfl_xor(s, off, 32);

    // argmax #1 (ties -> lowest index, matching jnp.argmax)
    float bv = (x0 >= x1) ? x0 : x1;
    int   bi = (x0 >= x1) ? i0 : i1;
    #pragma unroll
    for (int off = 16; off; off >>= 1) {
        float ov = __shfl_xor(bv, off, 32);
        int   oi = __shfl_xor(bi, off, 32);
        if (ov > bv || (ov == bv && oi < bi)) { bv = ov; bi = oi; }
    }
    // argmax #2 with top-1 masked out
    float y0 = (i0 == bi) ? -INFINITY : x0;
    float y1 = (i1 == bi) ? -INFINITY : x1;
    float cv = (y0 >= y1) ? y0 : y1;
    int   ci = (y0 >= y1) ? i0 : i1;
    #pragma unroll
    for (int off = 16; off; off >>= 1) {
        float ov = __shfl_xor(cv, off, 32);
        int   oi = __shfl_xor(ci, off, 32);
        if (ov > cv || (ov == cv && oi < ci)) { cv = ov; ci = oi; }
    }

    if (lane == 0) {
        const float inv = 1.0f / s;
        top1[tok] = bi;
        top2[tok] = ci;
        w1[tok] = __expf(bv - mx) * inv;   // prob of top-1 expert
        w2[tok] = __expf(cv - mx) * inv;   // prob of top-2 expert
    }
}

// ---------------------------------------------------------------------------
// K2: within-tile exclusive cumsum of one-hot masks via WMMA.
//   R[32x64] = StrictLowerTri[32x32] x OneHot[32x64]
// realized as 2 (row halves) x 4 (expert chunks) v_wmma_f32_16x16x32_f16,
// done for both the top-1 and top-2 masks (16 WMMAs per tile).
// One wave32 per tile -> EXEC is all ones at every WMMA (ISA requirement).
// ---------------------------------------------------------------------------
__global__ void rank_wmma_kernel(const int* __restrict__ top1,
                                 const int* __restrict__ top2,
                                 int* __restrict__ rin1, int* __restrict__ rin2,
                                 int* __restrict__ cnt1, int* __restrict__ cnt2) {
    __shared__ int   sT[2][32];
    __shared__ float R[32][NEXP];

    const int tile = blockIdx.x;
    const int lane = threadIdx.x;

    sT[0][lane] = top1[tile * 32 + lane];
    sT[1][lane] = top2[tile * 32 + lane];
    __syncthreads();

    // Build the two triangular A matrices once (A 16x32 in this code's contiguous k order, same as B:
    // lanes 0-15 hold K in 0..15, lanes 16-31 hold K in 16..31,
    // row M = lane % 16).
    const int mrow = lane & 15;
    const int khiA = (lane < 16) ? 0 : 16;
    v16h aLo, aHi;
    #pragma unroll
    for (int p = 0; p < 16; ++p) {
        const int k = p + khiA;
        aLo[p] = (_Float16)((k < mrow)      ? 1.0f : 0.0f);  // rows 0..15
        aHi[p] = (_Float16)((k < mrow + 16) ? 1.0f : 0.0f);  // rows 16..31
    }

    // B 32x16 (contiguous k, as in A): lanes 0-15 -> col N=lane, K=0..15; lanes 16-31 -> K=16..31.
    const int kbB  = (lane < 16) ? 0 : 16;
    const int ncol = lane & 15;

    for (int mat = 0; mat < 2; ++mat) {
        const int* st = sT[mat];
        #pragma unroll
        for (int c = 0; c < 4; ++c) {
            const int ecol = ncol + 16 * c;
            v16h b;
            #pragma unroll
            for (int p = 0; p < 16; ++p)
                b[p] = (_Float16)((st[kbB + p] == ecol) ? 1.0f : 0.0f);

            v8f dLo = {}, dHi = {};
            dLo = __builtin_amdgcn_wmma_f32_16x16x32_f16(
                      false, aLo, false, b, (short)0, dLo, false, false);
            dHi = __builtin_amdgcn_wmma_f32_16x16x32_f16(
                      false, aHi, false, b, (short)0, dHi, false, false);

            // C/D layout: VGPR j holds row j + 8*(lane>=16), col = lane%16.
            const int rbase = (lane < 16) ? 0 : 8;
            #pragma unroll
            for (int j = 0; j < 8; ++j) {
                R[rbase + j][ecol]      = dLo[j];
                R[16 + rbase + j][ecol] = dHi[j];
            }
        }
        __syncthreads();

        // rank-in-tile for this lane's token = exclusive count at its expert
        const int e = st[lane];
        const int rin = (int)R[lane][e];
        (mat ? rin2 : rin1)[tile * 32 + lane] = rin;

        // per-expert tile totals = exclusive count at row 31 + last token's hit
        for (int ee = lane; ee < NEXP; ee += 32) {
            const int tot = (int)R[31][ee] + ((st[31] == ee) ? 1 : 0);
            (mat ? cnt2 : cnt1)[tile * NEXP + ee] = tot;
        }
        __syncthreads();
    }
}

// ---------------------------------------------------------------------------
// K3: cross-tile exclusive scan per expert (64 threads, 256 tiles) +
//     used_capacity = min(CAP, c1 + c2).
// ---------------------------------------------------------------------------
__global__ void scan_kernel(const int* __restrict__ cnt1, const int* __restrict__ cnt2,
                            int* __restrict__ off1, int* __restrict__ off2,
                            int* __restrict__ c1tot, float* __restrict__ used_out) {
    const int e = threadIdx.x;   // 0..63
    int run = 0;
    for (int t = 0; t < NTILE; ++t) {
        off1[t * NEXP + e] = run;
        run += cnt1[t * NEXP + e];
    }
    const int c1 = run;
    c1tot[e] = c1;
    run = 0;
    for (int t = 0; t < NTILE; ++t) {
        off2[t * NEXP + e] = run;
        run += cnt2[t * NEXP + e];
    }
    const int c2 = run;
    int used = c1 + c2;
    if (used > CAP) used = CAP;
    used_out[e] = (float)used;
}

// ---------------------------------------------------------------------------
// K4: streaming zero-fill of cb_weight + sec_mask (1.34 GB) with 128-bit
//     non-temporal stores — this is the HBM-bandwidth-bound part.
// ---------------------------------------------------------------------------
__global__ void zero_kernel(vf4* __restrict__ p, long n4) {
    const long i = (long)blockIdx.x * blockDim.x + threadIdx.x;
    if (i < n4) {
        vf4 z = {0.0f, 0.0f, 0.0f, 0.0f};
        __builtin_nontemporal_store(z, p + i);
    }
}

// ---------------------------------------------------------------------------
// K5: sparse scatter — at most 2 slots per token survive capacity dropping.
// ---------------------------------------------------------------------------
__global__ void scatter_kernel(const int* __restrict__ top1, const int* __restrict__ top2,
                               const float* __restrict__ w1, const float* __restrict__ w2,
                               const int* __restrict__ rin1, const int* __restrict__ rin2,
                               const int* __restrict__ off1, const int* __restrict__ off2,
                               const int* __restrict__ c1tot,
                               float* __restrict__ cb, float* __restrict__ sec) {
    const int t    = blockIdx.x * blockDim.x + threadIdx.x;
    const int tile = t >> 5;

    const int e1 = top1[t];
    const int r1 = off1[tile * NEXP + e1] + rin1[t];
    if (r1 < CAP) {
        const size_t o = (size_t)t * (NEXP * CAP) + (size_t)e1 * CAP + r1;
        cb[o]  = w1[t];
        sec[o] = 1.0f;
    }
    const int e2 = top2[t];
    const int r2 = off2[tile * NEXP + e2] + rin2[t] + c1tot[e2];
    if (r2 < CAP) {
        const size_t o = (size_t)t * (NEXP * CAP) + (size_t)e2 * CAP + r2;
        cb[o]  = w2[t];
        sec[o] = 1.0f;
    }
}

// ---------------------------------------------------------------------------
extern "C" void kernel_launch(void* const* d_in, const int* in_sizes, int n_in,
                              void* d_out, int out_size, void* d_ws, size_t ws_size,
                              hipStream_t stream) {
    const float* x   = (const float*)d_in[0];
    float*       out = (float*)d_out;

    // d_out layout (reference return order, flattened):
    //   [0, 64)                      used_capacity
    //   [64, 64 + T*E*CAP)           cb_weight
    //   [64 + T*E*CAP, ... )         sec_mask
    float* used = out;
    float* cb   = out + NEXP;
    float* sec  = out + NEXP + (size_t)T_TOK * NEXP * CAP;

    // workspace carve-up
    char* w = (char*)d_ws;
    int*   top1  = (int*)w;              w += T_TOK * sizeof(int);
    int*   top2  = (int*)w;              w += T_TOK * sizeof(int);
    int*   rin1  = (int*)w;              w += T_TOK * sizeof(int);
    int*   rin2  = (int*)w;              w += T_TOK * sizeof(int);
    float* w1    = (float*)w;            w += T_TOK * sizeof(float);
    float* w2    = (float*)w;            w += T_TOK * sizeof(float);
    int*   cnt1  = (int*)w;              w += NTILE * NEXP * sizeof(int);
    int*   cnt2  = (int*)w;              w += NTILE * NEXP * sizeof(int);
    int*   off1  = (int*)w;              w += NTILE * NEXP * sizeof(int);
    int*   off2  = (int*)w;              w += NTILE * NEXP * sizeof(int);
    int*   c1tot = (int*)w;              w += NEXP * sizeof(int);

    // K1: one wave per token -> 8192 waves = 1024 blocks x 256 threads
    topk_kernel<<<T_TOK / 8, 256, 0, stream>>>(x, top1, top2, w1, w2);

    // K2: one wave32 per 32-token tile (EXEC all ones for WMMA)
    rank_wmma_kernel<<<NTILE, 32, 0, stream>>>(top1, top2, rin1, rin2, cnt1, cnt2);

    // K3: tiny per-expert scan over tiles
    scan_kernel<<<1, NEXP, 0, stream>>>(cnt1, cnt2, off1, off2, c1tot, used);

    // K4: zero cb_weight + sec_mask regions (2 * T*E*CAP floats, /4 per vf4)
    const long n4 = (2L * T_TOK * NEXP * CAP) / 4;            // 83,886,080
    const long zblocks = (n4 + 255) / 256;                    // 327,680
    zero_kernel<<<(unsigned)zblocks, 256, 0, stream>>>((vf4*)cb, n4);

    // K5: sparse scatter of surviving (token, expert, slot) entries
    scatter_kernel<<<T_TOK / 256, 256, 0, stream>>>(top1, top2, w1, w2,
                                                    rin1, rin2, off1, off2,
                                                    c1tot, cb, sec);
}
